// GreatNet_86174223827495
// MI455X (gfx1250) — matrix operand order repaired and checked by reading, not compiled, not run on hardware
//
#include <hip/hip_runtime.h>
#include <hip/hip_bf16.h>

// ---------------- types ----------------
typedef __attribute__((ext_vector_type(16))) __bf16          v16bf;
typedef __attribute__((ext_vector_type(16))) unsigned short  v16u;
typedef __attribute__((ext_vector_type(8)))  unsigned short  v8u;
typedef __attribute__((ext_vector_type(8)))  float           v8f;

#define N_AGT 100000
#define N_CTX 10000
#define N_EDGE 1000000
#define DCH 128

__device__ __forceinline__ unsigned short f2bf(float f) {
    unsigned int u = __builtin_bit_cast(unsigned int, f);
    u += 0x7FFFu + ((u >> 16) & 1u);           // round-to-nearest-even
    return (unsigned short)(u >> 16);
}

// ---------------- per-row GroupNorm stats over 128 channels ----------------
__device__ __forceinline__ void row_stats(const float* s_t, float* s_mean, float* s_rstd, int lane) {
    if (lane < 16) {
        const float4* r = reinterpret_cast<const float4*>(s_t + lane * DCH);
        float s = 0.f, ss = 0.f;
#pragma unroll
        for (int i = 0; i < 32; ++i) {
            float4 v = r[i];
            s  += v.x + v.y + v.z + v.w;
            ss += v.x * v.x + v.y * v.y + v.z * v.z + v.w * v.w;
        }
        float m   = s * (1.f / 128.f);
        float var = ss * (1.f / 128.f) - m * m;
        s_mean[lane] = m;
        s_rstd[lane] = rsqrtf(fmaxf(var, 0.f) + 1e-5f);
    }
}

// GN (+optional ReLU) fused into f32->bf16 requantization into an A tile.
__device__ __forceinline__ void gn_store_bf16(const float* s_t, unsigned short* s_dst, int dstride, int coff,
                                              const float* s_mean, const float* s_rstd,
                                              const float* __restrict__ gw, const float* __restrict__ gb,
                                              int lane, bool do_relu) {
    for (int i = lane; i < 16 * DCH; i += 32) {
        int row = i >> 7, col = i & 127;
        float v = (s_t[i] - s_mean[row]) * s_rstd[row];
        v = v * gw[col] + gb[col];
        if (do_relu) v = fmaxf(v, 0.f);
        s_dst[row * dstride + coff + col] = f2bf(v);
    }
}

// ---------------- 16xK @ KT-tiled WMMA -> 16x128 f32 tile in LDS ----------------
// s_a: bf16 A tile, 16 rows, row stride = astride (= KT*32) ushorts, in LDS.
// packB: B fragments pre-packed so each lane reads one contiguous 32B chunk.
template<int KT>
__device__ __forceinline__ void wmma_tile(const unsigned short* s_a, int astride,
                                          const unsigned short* __restrict__ packB,
                                          float* s_t, int lane) {
    const int row  = lane & 15;
    const int half = lane >> 4;
    v8f acc[8] = {};
#pragma unroll
    for (int kt = 0; kt < KT; ++kt) {
        // A 16x32 bf16 fragment: lane row = lane&15; K runs [half*8, half*8+7] and [16+half*8, ...]
        v16u a;
        v8u* ap = reinterpret_cast<v8u*>(&a);
        const unsigned short* abase = s_a + row * astride + kt * 32 + half * 8;
        ap[0] = *reinterpret_cast<const v8u*>(abase);        // ds_load_b128
        ap[1] = *reinterpret_cast<const v8u*>(abase + 16);   // ds_load_b128
        v16bf abf = __builtin_bit_cast(v16bf, a);
#pragma unroll
        for (int nt = 0; nt < 8; ++nt) {
            v16u braw = *reinterpret_cast<const v16u*>(packB + (((kt * 8 + nt) * 32 + lane) << 4)); // 32B global load
            v16bf bbf = __builtin_bit_cast(v16bf, braw);
            acc[nt] = __builtin_amdgcn_wmma_f32_16x16x32_bf16(
                false, abf, false, bbf, (short)0, acc[nt], false, false);
        }
    }
    // C/D layout: VGPR r -> M = r + 8*half, N = nt*16 + (lane&15)
#pragma unroll
    for (int nt = 0; nt < 8; ++nt)
#pragma unroll
        for (int r = 0; r < 8; ++r)
            s_t[(r + half * 8) * DCH + nt * 16 + row] = acc[nt][r];
}

// ---------------- prep kernels ----------------
// Pack weight W[N][K] (row-major f32, used as X @ W^T) into WMMA-B fragment order bf16.
__global__ void pack_b_kernel(const float* __restrict__ w, unsigned short* __restrict__ out, int N, int K) {
    int tid = blockIdx.x * blockDim.x + threadIdx.x;
    if (tid >= N * K) return;
    int e    = tid & 15;
    int lane = (tid >> 4) & 31;
    int frag = tid >> 9;
    int NT   = N >> 4;
    int nt   = frag % NT;
    int kt   = frag / NT;
    int k = kt * 32 + (lane >> 4) * 8 + (e & 7) + (e >> 3) * 16;   // B[k][n] = W[n][k]
    int n = nt * 16 + (lane & 15);
    out[tid] = f2bf(w[n * K + k]);
}

__global__ void cvt_bf16_kernel(const float* __restrict__ in, unsigned short* __restrict__ out, int n) {
    int i = blockIdx.x * blockDim.x + threadIdx.x;
    if (i < n) out[i] = f2bf(in[i]);
}

// ---------------- phase 1: per-agent GEMMs ----------------
// qa = relu(GN(agts @ q_w^T))  (bf16, gathered later per-edge)
// a0 = agts @ agt_w^T          (f32, scatter target)
__global__ void agent_pre_kernel(const float* __restrict__ agts,
                                 const unsigned short* __restrict__ packQ,
                                 const unsigned short* __restrict__ packA,
                                 const float* __restrict__ q_gw, const float* __restrict__ q_gb,
                                 unsigned short* __restrict__ qa, float* __restrict__ a0) {
    __shared__ __align__(16) unsigned short s_a[16 * DCH];
    __shared__ __align__(16) float s_t[16 * DCH];
    __shared__ float s_mean[16], s_rstd[16];
    const int  lane = threadIdx.x;
    const long r0   = (long)blockIdx.x * 16;

    const float4* src = reinterpret_cast<const float4*>(agts + r0 * DCH);
    for (int i = lane; i < 512; i += 32) {
        float4 v = src[i];
        int o = i * 4;
        s_a[o + 0] = f2bf(v.x); s_a[o + 1] = f2bf(v.y);
        s_a[o + 2] = f2bf(v.z); s_a[o + 3] = f2bf(v.w);
    }
    __syncthreads();

    wmma_tile<4>(s_a, DCH, packQ, s_t, lane);
    __syncthreads();
    row_stats(s_t, s_mean, s_rstd, lane);
    __syncthreads();
    for (int i = lane; i < 2048; i += 32) {
        int row = i >> 7, col = i & 127;
        float v = (s_t[i] - s_mean[row]) * s_rstd[row] * q_gw[col] + q_gb[col];
        qa[r0 * DCH + i] = f2bf(fmaxf(v, 0.f));
    }
    __syncthreads();

    wmma_tile<4>(s_a, DCH, packA, s_t, lane);
    __syncthreads();
    for (int i = lane; i < 2048; i += 32) a0[r0 * DCH + i] = s_t[i];
}

// ---------------- phase 2: per-edge fused pipeline (one wave = 16 edges) ----------------
__global__ void edge_kernel(const float* __restrict__ agt_ctrs, const float* __restrict__ ctx_ctrs,
                            const int* __restrict__ hi, const int* __restrict__ wi,
                            const float* __restrict__ dist_w1, const float* __restrict__ dist_b1,
                            const unsigned short* __restrict__ packD2,
                            const float* __restrict__ dist_gw, const float* __restrict__ dist_gb,
                            const unsigned short* __restrict__ qa, const unsigned short* __restrict__ ctxbf,
                            const unsigned short* __restrict__ packC1,
                            const float* __restrict__ ctx_gw, const float* __restrict__ ctx_gb,
                            const unsigned short* __restrict__ packC2,
                            float* __restrict__ a0, int E) {
    __shared__ __align__(16) unsigned short s_a[16 * 384];  // concat tile [d | q | ctx], stride 384
    __shared__ __align__(16) unsigned short s_d1[16 * DCH]; // dist-MLP hidden tile, stride 128
    __shared__ __align__(16) float s_t[16 * DCH];
    __shared__ float s_mean[16], s_rstd[16];
    __shared__ float s_d[32];
    __shared__ int s_h[16], s_w[16];
    const int  lane = threadIdx.x;
    const long e0   = (long)blockIdx.x * 16;

    if (lane < 16) {
        long e = e0 + lane; if (e >= E) e = E - 1;
        int h = hi[e], w = wi[e];
        s_h[lane] = h; s_w[lane] = w;
        s_d[lane * 2 + 0] = agt_ctrs[h * 2 + 0] - ctx_ctrs[w * 2 + 0];
        s_d[lane * 2 + 1] = agt_ctrs[h * 2 + 1] - ctx_ctrs[w * 2 + 1];
    }
    __syncthreads();

    // Issue the irregular gathers FIRST: qa[hi] -> concat cols [128,256),
    // ctx[wi] -> concat cols [256,384). Their L2 latency overlaps the whole
    // dist-MLP + dist_w2 WMMA chain below (they are only consumed at the
    // K=384 GEMM, four barriers later).
    for (int i = lane; i < 2048; i += 32) {
        int row = i >> 7, col = i & 127;
        s_a[row * 384 + 128 + col] = qa[(long)s_h[row] * DCH + col];
        s_a[row * 384 + 256 + col] = ctxbf[(long)s_w[row] * DCH + col];
    }

    // d1 = relu(d @ dist_w1^T + b1): tiny K=2, pure VALU -> bf16 tile (own buffer, stride 128)
    for (int i = lane; i < 2048; i += 32) {
        int row = i >> 7, col = i & 127;
        float v = dist_w1[col * 2 + 0] * s_d[row * 2 + 0]
                + dist_w1[col * 2 + 1] * s_d[row * 2 + 1] + dist_b1[col];
        s_d1[row * DCH + col] = f2bf(fmaxf(v, 0.f));
    }
    __syncthreads();

    // d2 = relu(GN(d1 @ dist_w2^T)) -> concat cols [0,128)
    wmma_tile<4>(s_d1, DCH, packD2, s_t, lane);
    __syncthreads();
    row_stats(s_t, s_mean, s_rstd, lane);
    __syncthreads();
    gn_store_bf16(s_t, s_a, 384, 0, s_mean, s_rstd, dist_gw, dist_gb, lane, true);
    __syncthreads();

    // c1 = relu(GN(concat @ ctx_w1^T)), K = 384
    wmma_tile<12>(s_a, 384, packC1, s_t, lane);
    __syncthreads();
    row_stats(s_t, s_mean, s_rstd, lane);
    __syncthreads();
    gn_store_bf16(s_t, s_d1, DCH, 0, s_mean, s_rstd, ctx_gw, ctx_gb, lane, true);
    __syncthreads();

    // c2 = c1 @ ctx_w2^T, then scatter-add into a0[hi]
    wmma_tile<4>(s_d1, DCH, packC2, s_t, lane);
    __syncthreads();
    for (int i = lane; i < 2048; i += 32) {
        int row = i >> 7;
        long e = e0 + row;
        if (e < E) atomicAdd(&a0[(long)s_h[row] * DCH + (i & 127)], s_t[i]);
    }
}

// ---------------- phase 3: per-agent epilogue ----------------
__global__ void agent_post_kernel(const float* __restrict__ a0, const float* __restrict__ agts,
                                  const float* __restrict__ norm_gw, const float* __restrict__ norm_gb,
                                  const unsigned short* __restrict__ packL,
                                  const float* __restrict__ lin_gw, const float* __restrict__ lin_gb,
                                  float* __restrict__ out) {
    __shared__ __align__(16) unsigned short s_a[16 * DCH];
    __shared__ __align__(16) float s_t[16 * DCH];
    __shared__ float s_mean[16], s_rstd[16];
    const int  lane = threadIdx.x;
    const long r0   = (long)blockIdx.x * 16;

    const float4* src = reinterpret_cast<const float4*>(a0 + r0 * DCH);
    float4* dstt = reinterpret_cast<float4*>(s_t);
    for (int i = lane; i < 512; i += 32) dstt[i] = src[i];
    __syncthreads();
    row_stats(s_t, s_mean, s_rstd, lane);
    __syncthreads();
    gn_store_bf16(s_t, s_a, DCH, 0, s_mean, s_rstd, norm_gw, norm_gb, lane, true);
    __syncthreads();

    wmma_tile<4>(s_a, DCH, packL, s_t, lane);
    __syncthreads();
    row_stats(s_t, s_mean, s_rstd, lane);
    __syncthreads();
    for (int i = lane; i < 2048; i += 32) {
        int row = i >> 7, col = i & 127;
        float v = (s_t[i] - s_mean[row]) * s_rstd[row] * lin_gw[col] + lin_gb[col];
        v += agts[r0 * DCH + i];
        out[r0 * DCH + i] = fmaxf(v, 0.f);
    }
}

// ---------------- launcher ----------------
extern "C" void kernel_launch(void* const* d_in, const int* in_sizes, int n_in,
                              void* d_out, int out_size, void* d_ws, size_t ws_size,
                              hipStream_t stream) {
    (void)in_sizes; (void)n_in; (void)out_size; (void)ws_size;
    const float* agts     = (const float*)d_in[0];
    const float* ctx      = (const float*)d_in[1];
    const float* agt_ctrs = (const float*)d_in[2];
    const float* ctx_ctrs = (const float*)d_in[3];
    const int*   hi       = (const int*)d_in[4];
    const int*   wi       = (const int*)d_in[5];
    const float* dist_w1  = (const float*)d_in[6];
    const float* dist_b1  = (const float*)d_in[7];
    const float* dist_w2  = (const float*)d_in[8];
    const float* dist_gw  = (const float*)d_in[9];
    const float* dist_gb  = (const float*)d_in[10];
    const float* q_w      = (const float*)d_in[11];
    const float* q_gw     = (const float*)d_in[12];
    const float* q_gb     = (const float*)d_in[13];
    const float* ctx_w1   = (const float*)d_in[14];
    const float* ctx_gw   = (const float*)d_in[15];
    const float* ctx_gb   = (const float*)d_in[16];
    const float* ctx_w2   = (const float*)d_in[17];
    const float* agt_w    = (const float*)d_in[18];
    const float* norm_gw  = (const float*)d_in[19];
    const float* norm_gb  = (const float*)d_in[20];
    const float* lin_w    = (const float*)d_in[21];
    const float* lin_gw   = (const float*)d_in[22];
    const float* lin_gb   = (const float*)d_in[23];
    float* out = (float*)d_out;

    // workspace carve-up (256B aligned)
    auto alignup = [](size_t x) { return (x + 255) & ~(size_t)255; };
    char* ws = (char*)d_ws;
    unsigned short* qa    = (unsigned short*)ws; ws += alignup((size_t)N_AGT * DCH * 2);
    unsigned short* ctxbf = (unsigned short*)ws; ws += alignup((size_t)N_CTX * DCH * 2);
    float*          a0    = (float*)ws;          ws += alignup((size_t)N_AGT * DCH * 4);
    unsigned short* packQ  = (unsigned short*)ws; ws += alignup((size_t)DCH * DCH * 2);
    unsigned short* packA  = (unsigned short*)ws; ws += alignup((size_t)DCH * DCH * 2);
    unsigned short* packD2 = (unsigned short*)ws; ws += alignup((size_t)DCH * DCH * 2);
    unsigned short* packC1 = (unsigned short*)ws; ws += alignup((size_t)DCH * 384 * 2);
    unsigned short* packC2 = (unsigned short*)ws; ws += alignup((size_t)DCH * DCH * 2);
    unsigned short* packL  = (unsigned short*)ws; ws += alignup((size_t)DCH * DCH * 2);

    // phase 0: weight packing + ctx bf16 conversion
    cvt_bf16_kernel<<<(N_CTX * DCH + 255) / 256, 256, 0, stream>>>(ctx, ctxbf, N_CTX * DCH);
    pack_b_kernel<<<(DCH * DCH) / 256, 256, 0, stream>>>(q_w,     packQ,  DCH, DCH);
    pack_b_kernel<<<(DCH * DCH) / 256, 256, 0, stream>>>(agt_w,   packA,  DCH, DCH);
    pack_b_kernel<<<(DCH * DCH) / 256, 256, 0, stream>>>(dist_w2, packD2, DCH, DCH);
    pack_b_kernel<<<(DCH * 384) / 256, 256, 0, stream>>>(ctx_w1,  packC1, DCH, 384);
    pack_b_kernel<<<(DCH * DCH) / 256, 256, 0, stream>>>(ctx_w2,  packC2, DCH, DCH);
    pack_b_kernel<<<(DCH * DCH) / 256, 256, 0, stream>>>(lin_w,   packL,  DCH, DCH);

    // phase 1: per-agent GEMMs (q branch + agt transform)
    agent_pre_kernel<<<N_AGT / 16, 32, 0, stream>>>(agts, packQ, packA, q_gw, q_gb, qa, a0);

    // phase 2: fused edge pipeline + scatter
    edge_kernel<<<N_EDGE / 16, 32, 0, stream>>>(agt_ctrs, ctx_ctrs, hi, wi,
                                                dist_w1, dist_b1, packD2, dist_gw, dist_gb,
                                                qa, ctxbf, packC1, ctx_gw, ctx_gb,
                                                packC2, a0, N_EDGE);

    // phase 3: epilogue GN + lin GEMM + residual
    agent_post_kernel<<<N_AGT / 16, 32, 0, stream>>>(a0, agts, norm_gw, norm_gb,
                                                     packL, lin_gw, lin_gb, out);
}
